// AttentionHead_52106543235875
// MI455X (gfx1250) — matrix operand order repaired and checked by reading, not compiled, not run on hardware
//
#include <hip/hip_runtime.h>

#define D_MODEL 1024
#define D_KEY   64
#define BATCH   4
#define SEQ     4096

typedef _Float16 f16;
typedef __attribute__((ext_vector_type(16))) _Float16 v16h;
typedef __attribute__((ext_vector_type(8)))  _Float16 v8h;
typedef __attribute__((ext_vector_type(8)))  float    v8f;
typedef __attribute__((ext_vector_type(4)))  float    v4f;

union V16 { v16h v; v8h h[2]; };

__device__ __forceinline__ v8f zero8() {
    v8f z;
#pragma unroll
    for (int i = 0; i < 8; ++i) z[i] = 0.0f;
    return z;
}

// ---- CDNA5 async global->LDS copy (ASYNCcnt-tracked), with sync fallback ----
#if defined(__has_builtin)
#if __has_builtin(__builtin_amdgcn_global_load_async_to_lds_b128)
#define HAS_ASYNC_LDS 1
#endif
#endif
#ifndef HAS_ASYNC_LDS
#define HAS_ASYNC_LDS 0
#endif

// Builtin signature (from hipcc diagnostic): param0 is
//   'int __attribute__((vector_size(16))) __device__ *'  i.e. v4i in AS(1).
typedef int v4i __attribute__((__vector_size__(16)));
typedef __attribute__((address_space(1))) v4i gv4i;
typedef __attribute__((address_space(3))) v4i lv4i;

__device__ __forceinline__ void async_cp16(const f16* gsrc, f16* lds_dst) {
#if HAS_ASYNC_LDS
    gv4i* g = (gv4i*)(unsigned long long)gsrc;
    // generic LDS address: low 32 bits are the LDS byte offset
    lv4i* l = (lv4i*)(unsigned)(unsigned long long)lds_dst;
    __builtin_amdgcn_global_load_async_to_lds_b128(g, l, 0, 0);
#else
    *(v8h*)lds_dst = *(const v8h*)gsrc;
#endif
}

__device__ __forceinline__ void wait_async0() {
#if HAS_ASYNC_LDS
#if __has_builtin(__builtin_amdgcn_s_wait_asynccnt)
    __builtin_amdgcn_s_wait_asynccnt(0);
#else
    asm volatile("s_wait_asynccnt 0x0" ::: "memory");
#endif
#endif
}

// ---------------------------------------------------------------------------
// Projection helpers: stage one 32xK, 64xN chunk of W transposed into LDS.
// Thread t: row kk = t>>3, cols n0..n0+7 -> 2 x b128 global, 8 x b16 LDS stores.
// ---------------------------------------------------------------------------
__device__ __forceinline__ void stage_w(const float* __restrict__ W, int k0,
                                        f16* Wcb, int tid)
{
    const int kk = tid >> 3;           // 0..31
    const int n0 = (tid & 7) * 8;      // 0..56
    const float* src = W + (size_t)(k0 + kk) * D_KEY + n0;
    const v4f w0 = *(const v4f*)src;
    const v4f w1 = *(const v4f*)(src + 4);
#pragma unroll
    for (int j = 0; j < 4; ++j) {
        Wcb[(n0 + j) * 32 + kk]     = (f16)w0[j];
        Wcb[(n0 + 4 + j) * 32 + kk] = (f16)w1[j];
    }
}

// ---------------------------------------------------------------------------
// Projection: Y = X @ W  (f32 in, f16 out), one blockIdx.y per {q,k,v}.
// q is pre-scaled by 1/sqrt(dk); v is written TRANSPOSED in global: vT[b][dk][seq].
// Double-buffered W chunk: staging overlaps WMMA, one barrier per K-step.
// ---------------------------------------------------------------------------
__global__ __launch_bounds__(256)
void proj_kernel(const float* __restrict__ Xq, const float* __restrict__ Xk,
                 const float* __restrict__ Xv, const float* __restrict__ Wq,
                 const float* __restrict__ Wk, const float* __restrict__ Wv,
                 f16* __restrict__ qf, f16* __restrict__ kf, f16* __restrict__ vfT)
{
    __shared__ f16 Wc[2][D_KEY * 32];  // transposed chunk: Wc[buf][n][kk]

    const int which = blockIdx.y;
    const float* X = (which == 0) ? Xq : (which == 1) ? Xk : Xv;
    const float* W = (which == 0) ? Wq : (which == 1) ? Wk : Wv;
    f16*        Y  = (which == 0) ? qf : (which == 1) ? kf : vfT;
    const float scale = (which == 0) ? 0.125f : 1.0f;

    const int tid  = threadIdx.x;
    const int w    = tid >> 5;
    const int lane = tid & 31;
    const int m    = lane & 15;
    const int h    = lane >> 4;
    const int r0   = blockIdx.x * 128 + w * 16;

    v8f acc[4];
#pragma unroll
    for (int t = 0; t < 4; ++t) acc[t] = zero8();

    stage_w(W, 0, Wc[0], tid);                 // prologue chunk

    for (int k0 = 0; k0 < D_MODEL; k0 += 32) {
        const int cb = (k0 >> 5) & 1;
        __syncthreads();                       // chunk cb fully staged
        if (k0 + 32 < D_MODEL)
            stage_w(W, k0 + 32, Wc[cb ^ 1], tid);   // overlaps WMMA below

        // A fragment (16x32 f16): lane row m, K(e) = e + 16h
        const float* arow = X + (size_t)(r0 + m) * D_MODEL + k0 + 16 * h;
        if (k0 + 32 < D_MODEL) __builtin_prefetch(arow + 32, 0, 1);
        const v4f a0 = *(const v4f*)(arow);
        const v4f a1 = *(const v4f*)(arow + 4);
        const v4f a2 = *(const v4f*)(arow + 8);
        const v4f a3 = *(const v4f*)(arow + 12);
        V16 A;
#pragma unroll
        for (int i = 0; i < 4; ++i) {
            A.v[i]      = (f16)a0[i];
            A.v[4 + i]  = (f16)a1[i];
            A.v[8 + i]  = (f16)a2[i];
            A.v[12 + i] = (f16)a3[i];
        }
#pragma unroll
        for (int t = 0; t < 4; ++t) {
            const v16h bf = *(const v16h*)&Wc[cb][(t * 16 + m) * 32 + 16 * h];
            acc[t] = __builtin_amdgcn_wmma_f32_16x16x32_f16(
                false, A.v, false, bf, (short)0, acc[t], false, false);
        }
    }

#pragma unroll
    for (int t = 0; t < 4; ++t)
#pragma unroll
        for (int r = 0; r < 8; ++r) {
            const int grow = r0 + r + 8 * h;           // b*SEQ + s
            const f16 val  = (f16)(acc[t][r] * scale);
            if (which == 2) {                          // V: transposed [b][dk][seq]
                const int bb = grow >> 12;
                const int s  = grow & (SEQ - 1);
                Y[((size_t)bb * D_KEY + t * 16 + m) * SEQ + s] = val;
            } else {
                Y[(size_t)grow * D_KEY + t * 16 + m] = val;
            }
        }
}

// ---------------------------------------------------------------------------
// Stage one 64-key K tile (row-major [key][dk]) and V tile ([dk][key], V already
// transposed in global) into LDS: 4 x b128 async copies per thread.
// ---------------------------------------------------------------------------
__device__ __forceinline__ void stage_tiles(const f16* kfb, const f16* vfb, int kt0,
                                            f16* ktb, f16* vtb, int tid)
{
    const int idx = tid * 16;        // 256*16 = 4096 = 64*64
    const int r64 = idx >> 6;        // tile row
    const int c16 = idx & 63;        // tile col base (multiple of 16)
    const f16* ks = kfb + (size_t)(kt0 + r64) * D_KEY + c16;   // K[key][dk]
    const f16* vs = vfb + (size_t)r64 * SEQ + kt0 + c16;       // vT[dk][seq]
    async_cp16(ks,     ktb + idx);
    async_cp16(ks + 8, ktb + idx + 8);
    async_cp16(vs,     vtb + idx);
    async_cp16(vs + 8, vtb + idx + 8);
}

// ---------------------------------------------------------------------------
// Flash attention, causal. 8 waves x 16 q-rows; double-buffered async K/V tiles.
// ---------------------------------------------------------------------------
__global__ __launch_bounds__(256)
void attn_kernel(const f16* __restrict__ qf, const f16* __restrict__ kf,
                 const f16* __restrict__ vfT, float* __restrict__ out)
{
    __shared__ f16 kt[2][64 * 64];   // [buf][key][dk]
    __shared__ f16 vt[2][64 * 64];   // [buf][dk][key]
    __shared__ f16 ps[8][16 * 32];   // per-wave P scratch [qrow][key]

    const int b    = blockIdx.y;
    const int qb   = blockIdx.x * 128;
    const int tid  = threadIdx.x;
    const int w    = tid >> 5;
    const int lane = tid & 31;
    const int m    = lane & 15;
    const int h    = lane >> 4;
    const int qr0  = qb + w * 16;

    const f16* kfb = kf  + (size_t)b * SEQ * D_KEY;
    const f16* vfb = vfT + (size_t)b * D_KEY * SEQ;

    // Q fragments (pre-scaled): dk = 2 x K32
    const f16* qrow = qf + ((size_t)b * SEQ + qr0 + m) * D_KEY;
    V16 Qa[2];
#pragma unroll
    for (int c = 0; c < 2; ++c) {
        Qa[c].h[0] = *(const v8h*)(qrow + 32 * c + 16 * h);
        Qa[c].h[1] = *(const v8h*)(qrow + 32 * c + 16 * h + 8);
    }

    float mrow[8], lrow[8];
#pragma unroll
    for (int r = 0; r < 8; ++r) { mrow[r] = -1e30f; lrow[r] = 0.0f; }
    v8f o[4];
#pragma unroll
    for (int d = 0; d < 4; ++d) o[d] = zero8();

    const int kmax = qb + 128;
    stage_tiles(kfb, vfb, 0, kt[0], vt[0], tid);     // prologue: tile 0 in flight

    for (int kt0 = 0; kt0 < kmax; kt0 += 64) {
        const int bf = (kt0 >> 6) & 1;
        wait_async0();            // own async copies for tile kt0 are done
        __syncthreads();          // => everyone's copies are done
        if (kt0 + 64 < kmax)      // overlap: next tile streams in during compute
            stage_tiles(kfb, vfb, kt0 + 64, kt[bf ^ 1], vt[bf ^ 1], tid);

#pragma unroll
        for (int sb = 0; sb < 2; ++sb) {
            const int kb = kt0 + sb * 32;            // wave-uniform
            if (kb > qr0 + 15) continue;             // fully masked for this wave
            const int kl = sb * 32;

            // S = Q K^T
            v8f s[2] = { zero8(), zero8() };
#pragma unroll
            for (int t = 0; t < 2; ++t)
#pragma unroll
                for (int c = 0; c < 2; ++c) {
                    const v16h bfr = *(const v16h*)&kt[bf][(kl + 16 * t + m) * D_KEY + 32 * c + 16 * h];
                    s[t] = __builtin_amdgcn_wmma_f32_16x16x32_f16(
                        false, Qa[c].v, false, bfr, (short)0, s[t], false, false);
                }

            // causal mask (arithmetic; EXEC stays all-ones around WMMA)
#pragma unroll
            for (int t = 0; t < 2; ++t) {
                const int key = kb + 16 * t + m;
#pragma unroll
                for (int r = 0; r < 8; ++r)
                    if (key > qr0 + r + 8 * h) s[t][r] = -1e30f;
            }

            // online softmax: row (r+8h) lives in one 16-lane half
            float red[8];
#pragma unroll
            for (int r = 0; r < 8; ++r) red[r] = fmaxf(s[0][r], s[1][r]);
#pragma unroll
            for (int off = 1; off < 16; off <<= 1)
#pragma unroll
                for (int r = 0; r < 8; ++r)
                    red[r] = fmaxf(red[r], __shfl_xor(red[r], off, 32));
#pragma unroll
            for (int r = 0; r < 8; ++r) {
                const float mn = fmaxf(mrow[r], red[r]);
                const float al = __expf(mrow[r] - mn);
                mrow[r] = mn;
                lrow[r] *= al;
#pragma unroll
                for (int d = 0; d < 4; ++d) o[d][r] *= al;
            }

            // P = exp(S-m) -> LDS [row][key] so it can be reloaded in A-layout
            f16* myp = &ps[w][0];
#pragma unroll
            for (int t = 0; t < 2; ++t)
#pragma unroll
                for (int r = 0; r < 8; ++r) {
                    const float p = __expf(s[t][r] - mrow[r]);
                    s[t][r] = p;
                    myp[(r + 8 * h) * 32 + 16 * t + m] = (f16)p;
                }
#pragma unroll
            for (int r = 0; r < 8; ++r) red[r] = s[0][r] + s[1][r];
#pragma unroll
            for (int off = 1; off < 16; off <<= 1)
#pragma unroll
                for (int r = 0; r < 8; ++r)
                    red[r] += __shfl_xor(red[r], off, 32);
#pragma unroll
            for (int r = 0; r < 8; ++r) lrow[r] += red[r];

            // intra-wave cross-lane LDS RAW: drain DScnt before A-layout reload
            asm volatile("s_wait_dscnt 0" ::: "memory");

            V16 Pa;
            Pa.h[0] = *(const v8h*)&myp[m * 32 + 16 * h];
            Pa.h[1] = *(const v8h*)&myp[m * 32 + 16 * h + 8];

            // O += P V
#pragma unroll
            for (int d = 0; d < 4; ++d) {
                const v16h vb = *(const v16h*)&vt[bf][(16 * d + m) * 64 + kl + 16 * h];
                o[d] = __builtin_amdgcn_wmma_f32_16x16x32_f16(
                    false, Pa.v, false, vb, (short)0, o[d], false, false);
            }
        }
    }

#pragma unroll
    for (int d = 0; d < 4; ++d)
#pragma unroll
        for (int r = 0; r < 8; ++r) {
            const int row = qr0 + r + 8 * h;
            out[((size_t)b * SEQ + row) * D_KEY + 16 * d + m] = o[d][r] / lrow[r];
        }
}

// ---------------------------------------------------------------------------
extern "C" void kernel_launch(void* const* d_in, const int* in_sizes, int n_in,
                              void* d_out, int out_size, void* d_ws, size_t ws_size,
                              hipStream_t stream) {
    const float* Q  = (const float*)d_in[0];
    const float* K  = (const float*)d_in[1];
    const float* V  = (const float*)d_in[2];
    // d_in[3] = causal mask (implied by kernel, unused)
    const float* Wq = (const float*)d_in[4];
    const float* Wk = (const float*)d_in[5];
    const float* Wv = (const float*)d_in[6];
    float* out = (float*)d_out;

    f16* qf  = (f16*)d_ws;
    f16* kf  = qf + (size_t)BATCH * SEQ * D_KEY;
    f16* vfT = kf + (size_t)BATCH * SEQ * D_KEY;

    dim3 gProj((BATCH * SEQ) / 128, 3);
    proj_kernel<<<gProj, 256, 0, stream>>>(Q, K, V, Wq, Wk, Wv, qf, kf, vfT);

    dim3 gAttn(SEQ / 128, BATCH);
    attn_kernel<<<gAttn, 256, 0, stream>>>(qf, kf, vfT, out);
}
